// TorchGroupGemmReduceRS_54563264529073
// MI455X (gfx1250) — matrix operand order repaired and checked by reading, not compiled, not run on hardware
//
#include <hip/hip_runtime.h>
#include <hip/hip_bf16.h>

// ---------------------------------------------------------------------------
// Problem constants (from reference)
// ---------------------------------------------------------------------------
#define T_TOK   16384   // num_tokens * topk
#define D_DIM   2048    // K of the GEMM
#define H_DIM   4096    // N of the GEMM
#define N_EXP   8

// GEMM tiling: block = 256 thr (8 wave32, grid 4M x 2N), tile 64M x 128N x 32K
#define MB      64
#define NB      128
#define KC      32
#define NCHUNK  (D_DIM / KC)                 // 64

// LDS rows are 5 blocks of 16B (4 data blocks = 32 bf16 + 1 rotation slot).
// Physical block = (logical block + row) % 5 -> kills bank conflicts for both
// the strided transposed stores and the 16B fragment loads.
#define ROW_BYTES 80
#define AH_OFF  0
#define AL_OFF  (MB * ROW_BYTES)             // 5120
#define BH_OFF  (2 * MB * ROW_BYTES)         // 10240
#define BL_OFF  (BH_OFF + NB * ROW_BYTES)    // 20480
#define BUF_BYTES (BL_OFF + NB * ROW_BYTES)  // 30720 ; x2 buffers = 60 KB

#define MAX_MTILES (T_TOK / MB + N_EXP)      // 264 worst case

typedef __attribute__((ext_vector_type(8)))  float   v8f;
typedef __attribute__((ext_vector_type(16))) __bf16  v16bf;
typedef __attribute__((ext_vector_type(8)))  __bf16  v8bf;

static __device__ __forceinline__ int rot5(int b, int r) {
    return (int)(((unsigned)(b + r)) % 5u) * 16;
}
static __device__ __forceinline__ void split_bf16(float x, __bf16& h, __bf16& l) {
    h = (__bf16)x;                 // RNE to bf16
    l = (__bf16)(x - (float)h);    // residual -> ~2^-16 combined accuracy
}
static __device__ __forceinline__ unsigned int pk2(__bf16 a, __bf16 b) {
    return (unsigned int)__builtin_bit_cast(unsigned short, a) |
           ((unsigned int)__builtin_bit_cast(unsigned short, b) << 16);
}

// ---------------------------------------------------------------------------
// Kernel 1: zero the output (we accumulate into it with atomics)
// ---------------------------------------------------------------------------
__global__ void zero_out_kernel(float4* __restrict__ p, int n4) {
    int i = blockIdx.x * blockDim.x + threadIdx.x;
    if (i < n4) p[i] = make_float4(0.f, 0.f, 0.f, 0.f);
}

// ---------------------------------------------------------------------------
// Kernel 2: deterministic stable sort-by-expert + tile table (one workgroup,
// per-expert Hillis-Steele scan in LDS; no atomics -> bit-deterministic).
// ---------------------------------------------------------------------------
__global__ __launch_bounds__(1024)
void route_kernel(const int* __restrict__ ids,
                  int* __restrict__ nmt,       // [1]
                  int* __restrict__ off,       // [N_EXP+1]
                  int* __restrict__ te,        // [MAX_MTILES]
                  int* __restrict__ tr,        // [MAX_MTILES]
                  int* __restrict__ sorted)    // [T_TOK]
{
    __shared__ int cnt[N_EXP * 1024];
    __shared__ int base[N_EXP + 1];
    const int t = threadIdx.x;
    const int tok0 = t * (T_TOK / 1024);

    int local[N_EXP];
#pragma unroll
    for (int e = 0; e < N_EXP; ++e) local[e] = 0;
    for (int i = 0; i < T_TOK / 1024; ++i) local[ids[tok0 + i]]++;
#pragma unroll
    for (int e = 0; e < N_EXP; ++e) cnt[e * 1024 + t] = local[e];
    __syncthreads();

    for (int s = 1; s < 1024; s <<= 1) {
        int v[N_EXP];
#pragma unroll
        for (int e = 0; e < N_EXP; ++e)
            v[e] = (t >= s) ? cnt[e * 1024 + t - s] : 0;
        __syncthreads();
#pragma unroll
        for (int e = 0; e < N_EXP; ++e) cnt[e * 1024 + t] += v[e];
        __syncthreads();
    }

    if (t == 0) {
        base[0] = 0;
        for (int e = 0; e < N_EXP; ++e) base[e + 1] = base[e] + cnt[e * 1024 + 1023];
        for (int e = 0; e <= N_EXP; ++e) off[e] = base[e];
        int nm = 0;
        for (int e = 0; e < N_EXP; ++e) {
            int c = base[e + 1] - base[e];
            for (int m = 0; m < c; m += MB) { te[nm] = e; tr[nm] = base[e] + m; nm++; }
        }
        *nmt = nm;
    }
    __syncthreads();

    int pos[N_EXP];
#pragma unroll
    for (int e = 0; e < N_EXP; ++e)
        pos[e] = base[e] + cnt[e * 1024 + t] - local[e];
    for (int i = 0; i < T_TOK / 1024; ++i) {
        int tok = tok0 + i;
        sorted[pos[ids[tok]]++] = tok;
    }
}

// ---------------------------------------------------------------------------
// Kernel 3: grouped GEMM, split-bf16 WMMA (hh+hl+lh), double-buffered LDS.
// ---------------------------------------------------------------------------
__global__ __launch_bounds__(256)
void gemm_kernel(const float* __restrict__ x,
                 const float* __restrict__ w,
                 const float* __restrict__ tkw,
                 const int* __restrict__ nmt,
                 const int* __restrict__ off,
                 const int* __restrict__ te,
                 const int* __restrict__ tr,
                 const int* __restrict__ sorted,
                 float* __restrict__ out)
{
    __shared__ __attribute__((aligned(16))) unsigned char lds[2 * BUF_BYTES];
    __shared__ int   rid[MB];
    __shared__ float rw[MB];

    const int bx = blockIdx.x;
    if (bx >= *nmt) return;                   // uniform across block

    const int e  = te[bx];
    const int r0 = tr[bx];
    int nrows = off[e + 1] - r0;
    if (nrows > MB) nrows = MB;

    const int tid = threadIdx.x;
    if (tid < MB) {
        int s = (tid < nrows) ? sorted[r0 + tid] : -1;
        rid[tid] = s;
        rw[tid]  = (s >= 0) ? tkw[s] : 0.f;
    }

    const float* we = w + (size_t)e * D_DIM * H_DIM;
    const int n0 = blockIdx.y * NB;

    const int wv   = tid >> 5;
    const int lane = tid & 31;
    const int wm   = wv >> 1;        // 0..3 (M wave)
    const int wn   = wv & 1;         // 0..1 (N wave)
    const int l2   = lane >> 4;
    const int mr   = lane & 15;

    // --- staging assignments (loop invariant) ---
    const int am = tid >> 2;         // 0..63 : A row
    const int ab = tid & 3;          // 0..3  : A 16B block (k = 8*ab..8*ab+7)
    const int bn = (tid & 31) * 4;   // 0..124: B col group
    const int kp = tid >> 5;         // 0..7  : B k quad (k = 4*kp..4*kp+3)

    // --- precomputed swizzled LDS byte offsets ---
    const int aSt = am * ROW_BYTES + rot5(ab, am);
    int bSt[4];
#pragma unroll
    for (int j = 0; j < 4; ++j)
        bSt[j] = (bn + j) * ROW_BYTES + rot5(kp >> 1, bn + j) + (kp & 1) * 8;

    const int Arow = wm * 16 + mr;
    const int aLd0 = Arow * ROW_BYTES + rot5(l2, Arow);
    const int aLd1 = Arow * ROW_BYTES + rot5(2 + l2, Arow);
    int bLd0[4], bLd1[4];
#pragma unroll
    for (int j = 0; j < 4; ++j) {
        int N = wn * 64 + j * 16 + mr;
        bLd0[j] = N * ROW_BYTES + rot5(l2, N);
        bLd1[j] = N * ROW_BYTES + rot5(2 + l2, N);
    }

    __syncthreads();                 // rid/rw visible

    const int aslot = rid[am];
    const float* aP = x + (size_t)(aslot < 0 ? 0 : aslot) * D_DIM + ab * 8;
    const float* bP = we + (size_t)(4 * kp) * H_DIM + n0 + bn;

    v8f acc[4];
#pragma unroll
    for (int j = 0; j < 4; ++j) acc[j] = (v8f)0.f;

    float4 aR0, aR1, bR0, bR1, bR2, bR3;

    // ---- global prefetch of one K-chunk into registers ----
    auto load_chunk = [&](int k0) {
        if (aslot >= 0) {
            aR0 = *(const float4*)(aP + k0);
            aR1 = *(const float4*)(aP + k0 + 4);
        } else {
            aR0 = make_float4(0.f, 0.f, 0.f, 0.f);
            aR1 = aR0;
        }
        bR0 = *(const float4*)(bP + (size_t)(k0 + 0) * H_DIM);
        bR1 = *(const float4*)(bP + (size_t)(k0 + 1) * H_DIM);
        bR2 = *(const float4*)(bP + (size_t)(k0 + 2) * H_DIM);
        bR3 = *(const float4*)(bP + (size_t)(k0 + 3) * H_DIM);
    };
    // ---- convert to hi/lo bf16 and store one chunk to an LDS buffer ----
    auto store_chunk = [&](unsigned char* dst) {
        // A: 8 consecutive k -> one b128 store each for hi and lo
        const float* av = &aR0.x;   // aR0,aR1 contiguous? not guaranteed: handle both
        float a8[8] = {aR0.x, aR0.y, aR0.z, aR0.w, aR1.x, aR1.y, aR1.z, aR1.w};
        __bf16 h[8], l[8];
#pragma unroll
        for (int q = 0; q < 8; ++q) split_bf16(a8[q], h[q], l[q]);
        uint4 hv = make_uint4(pk2(h[0], h[1]), pk2(h[2], h[3]), pk2(h[4], h[5]), pk2(h[6], h[7]));
        uint4 lv = make_uint4(pk2(l[0], l[1]), pk2(l[2], l[3]), pk2(l[4], l[5]), pk2(l[6], l[7]));
        *(uint4*)(dst + AH_OFF + aSt) = hv;
        *(uint4*)(dst + AL_OFF + aSt) = lv;
        (void)av;
        // B: per column j, pack the 4-k run transposed -> one b64 store per buf
#pragma unroll
        for (int j = 0; j < 4; ++j) {
            float f0 = (&bR0.x)[j], f1 = (&bR1.x)[j], f2 = (&bR2.x)[j], f3 = (&bR3.x)[j];
            __bf16 h0, h1, h2, h3, l0, l1, l2v, l3;
            split_bf16(f0, h0, l0); split_bf16(f1, h1, l1);
            split_bf16(f2, h2, l2v); split_bf16(f3, h3, l3);
            *(uint2*)(dst + BH_OFF + bSt[j]) = make_uint2(pk2(h0, h1), pk2(h2, h3));
            *(uint2*)(dst + BL_OFF + bSt[j]) = make_uint2(pk2(l0, l1), pk2(l2v, l3));
        }
    };

    load_chunk(0);
    store_chunk(lds);
    __syncthreads();

    int cur = 0;
    for (int i = 0; i < NCHUNK; ++i) {
        if (i + 1 < NCHUNK) load_chunk((i + 1) * KC);   // HBM latency hides behind WMMAs

        const unsigned char* B = lds + cur * BUF_BYTES;
        v16bf ah, al;
        ((v8bf*)&ah)[0] = *(const v8bf*)(B + AH_OFF + aLd0);
        ((v8bf*)&ah)[1] = *(const v8bf*)(B + AH_OFF + aLd1);
        ((v8bf*)&al)[0] = *(const v8bf*)(B + AL_OFF + aLd0);
        ((v8bf*)&al)[1] = *(const v8bf*)(B + AL_OFF + aLd1);
#pragma unroll
        for (int j = 0; j < 4; ++j) {
            v16bf bh, bl;
            ((v8bf*)&bh)[0] = *(const v8bf*)(B + BH_OFF + bLd0[j]);
            ((v8bf*)&bh)[1] = *(const v8bf*)(B + BH_OFF + bLd1[j]);
            ((v8bf*)&bl)[0] = *(const v8bf*)(B + BL_OFF + bLd0[j]);
            ((v8bf*)&bl)[1] = *(const v8bf*)(B + BL_OFF + bLd1[j]);

            acc[j] = __builtin_amdgcn_wmma_f32_16x16x32_bf16(
                false, ah, false, bh, (short)0, acc[j], false, false);
            acc[j] = __builtin_amdgcn_wmma_f32_16x16x32_bf16(
                false, ah, false, bl, (short)0, acc[j], false, false);
            acc[j] = __builtin_amdgcn_wmma_f32_16x16x32_bf16(
                false, al, false, bh, (short)0, acc[j], false, false);
        }

        if (i + 1 < NCHUNK) store_chunk(lds + (cur ^ 1) * BUF_BYTES);
        __syncthreads();
        cur ^= 1;
    }

    // ---- epilogue: scale by slot weight, atomic-add into out[slot>>1] ----
    // C layout: VGPR r -> M = r + 8*l2, N = lane&15
#pragma unroll
    for (int j = 0; j < 4; ++j) {
        int col = n0 + wn * 64 + j * 16 + mr;
#pragma unroll
        for (int r = 0; r < 8; ++r) {
            int M = wm * 16 + l2 * 8 + r;
            int slot = rid[M];
            if (slot >= 0) {
                float val = acc[j][r] * rw[M];
                atomicAdd(out + (size_t)(slot >> 1) * H_DIM + col, val);
            }
        }
    }
}

// ---------------------------------------------------------------------------
// Launch
// ---------------------------------------------------------------------------
extern "C" void kernel_launch(void* const* d_in, const int* in_sizes, int n_in,
                              void* d_out, int out_size, void* d_ws, size_t ws_size,
                              hipStream_t stream) {
    const float* x   = (const float*)d_in[0];   // [T, D]
    const float* w   = (const float*)d_in[1];   // [E, D, H]
    const float* tkw = (const float*)d_in[2];   // [T]
    const int*   ids = (const int*)d_in[3];     // [T]
    float* out = (float*)d_out;                 // [T/2, H]

    // ws ints: [0]=n_mtiles, [1..9]=offsets, [16..]=tile_expert, tile_row0,
    // sorted_ids  (~67 KB total)
    int* nmt    = (int*)d_ws;
    int* off    = nmt + 1;
    int* te     = nmt + 16;
    int* tr     = te + MAX_MTILES;
    int* sorted = tr + MAX_MTILES;

    {
        int n4 = out_size / 4;
        int blocks = (n4 + 255) / 256;
        zero_out_kernel<<<blocks, 256, 0, stream>>>((float4*)out, n4);
    }
    route_kernel<<<1, 1024, 0, stream>>>(ids, nmt, off, te, tr, sorted);
    dim3 grid(MAX_MTILES, H_DIM / NB);
    gemm_kernel<<<grid, 256, 0, stream>>>(x, w, tkw, nmt, off, te, tr, sorted, out);
}
